// Block_25391846654035
// MI455X (gfx1250) — matrix operand order repaired and checked by reading, not compiled, not run on hardware
//
#include <hip/hip_runtime.h>
#include <hip/hip_bf16.h>

typedef __bf16 bf16;
typedef __attribute__((ext_vector_type(16))) __bf16 v16bf;
typedef __attribute__((ext_vector_type(8)))  float  v8f;
typedef __attribute__((ext_vector_type(4)))  unsigned int v4u;
typedef __attribute__((ext_vector_type(8)))  int          v8i;
typedef __attribute__((ext_vector_type(4)))  int          v4i;

#ifndef __has_builtin
#define __has_builtin(x) 0
#endif

#if __has_builtin(__builtin_amdgcn_global_load_async_to_lds_b128)
#define HAS_ASYNC 1
#else
#define HAS_ASYNC 0
#endif
#if __has_builtin(__builtin_amdgcn_tensor_load_to_lds)
#define HAS_TDM 1
#else
#define HAS_TDM 0
#endif
#if __has_builtin(__builtin_amdgcn_s_wait_asynccnt)
#define HAS_WAIT_ASYNC 1
#else
#define HAS_WAIT_ASYNC 0
#endif

#define Bb 8
#define Tt 2048
#define Cc 768
#define Hh 3072
#define Mm (Bb*Tt)

#define TM 128
#define TN 64
#define KC 32
#define LDS_STRIDE 40   // KC + 8 pad (bf16 elems): 64B row + 16B pad

union Frag { uint4 u[2]; v16bf v; };

#if HAS_ASYNC
// async b128 builtin params: (AS1 v4i* global_src, AS3 v4i* lds_dst, imm offset, imm cpol)
typedef __attribute__((address_space(1))) v4i as1_v4i;
typedef __attribute__((address_space(3))) v4i as3_v4i;
#endif

// byte offset of a __shared__ pointer within LDS
__device__ inline unsigned lds_off(const void* p) {
#if defined(__HIP_DEVICE_COMPILE__) && (HAS_ASYNC || HAS_TDM)
    return (unsigned)(size_t)(__attribute__((address_space(3))) const void*)p;
#else
    (void)p; return 0;
#endif
}

__device__ inline void wait_async0() {
#if HAS_WAIT_ASYNC
    __builtin_amdgcn_s_wait_asynccnt(0);
#elif HAS_ASYNC
    asm volatile("s_wait_asynccnt 0x0" ::: "memory");
#endif
}

#if HAS_TDM
// TDM 2D tile load: rows x KC (bf16) from row-major [tensor_rows][Kelems],
// LDS padded 16B after every 64B row -> LDS_STRIDE=40 bf16 row pitch.
__device__ inline void tdm_load_2d(const bf16* gsrc, unsigned lds_byte_off,
                                   int rows, int tensor_rows, int Kelems) {
    unsigned long long ga = (unsigned long long)gsrc;
    v4u g0 = { 1u,                                   // count=1, user desc
               lds_byte_off,                         // lds_addr
               (unsigned)ga,                         // global_addr[31:0]
               (unsigned)((ga >> 32) & 0x1FFFFFFull) | (2u << 30) }; // [56:32] | type=2
    const int w0 = (1 << 16)     // data_size = 2 bytes
                 | (1 << 20)     // pad_enable
                 | (3 << 22)     // pad_interval: 16 DWORDs (=64B, one tile row)
                 | (3 << 25);    // pad_amount:   4 DWORDs (=16B)
    v8i g1 = { w0,
               (int)((Kelems & 0xFFFF) << 16),                              // tensor_dim0 lo
               (int)(((Kelems >> 16) & 0xFFFF) | ((tensor_rows & 0xFFFF) << 16)), // dim0 hi|dim1 lo
               (int)(((tensor_rows >> 16) & 0xFFFF) | (KC << 16)),          // dim1 hi | tile_dim0=32
               rows,                                                         // tile_dim1 | tile_dim2=0
               Kelems,                                                       // tensor_dim0_stride lo
               0, 0 };
    v4i z4 = { 0, 0, 0, 0 };
    v8i z8 = { 0, 0, 0, 0, 0, 0, 0, 0 };
    __builtin_amdgcn_tensor_load_to_lds(g0, g1, z4, z4, z8, 0);
}
#endif

// ---------------------------------------------------------------------------
// Weight pack: f32 [K][N] row-major -> bf16 transposed [N][K]
// ---------------------------------------------------------------------------
__global__ void transpose_pack_bf16(const float* __restrict__ in,
                                    bf16* __restrict__ out, int K, int N) {
    int g = blockIdx.x * blockDim.x + threadIdx.x;
    if (g >= K * N) return;
    int n = g / K, k = g - n * K;
    out[(size_t)n * K + k] = (bf16)in[(size_t)k * N + n];
}

// ---------------------------------------------------------------------------
// LayerNorm over last dim (C=768), one block per row, 256 threads
// ---------------------------------------------------------------------------
__global__ __launch_bounds__(256)
void ln_kernel(const float* __restrict__ x, const float* __restrict__ w,
               const float* __restrict__ b, float* __restrict__ out) {
    const int row = blockIdx.x;
    const float* xr = x + (size_t)row * Cc;
    __shared__ float warp_part[8];
    __shared__ float stat[2];

    float s = 0.f;
    for (int i = threadIdx.x; i < Cc; i += 256) s += xr[i];
    #pragma unroll
    for (int o = 16; o > 0; o >>= 1) s += __shfl_down(s, o, 32);
    int wid = threadIdx.x >> 5, lane = threadIdx.x & 31;
    if (lane == 0) warp_part[wid] = s;
    __syncthreads();
    if (threadIdx.x == 0) {
        float t = 0.f;
        #pragma unroll
        for (int i = 0; i < 8; i++) t += warp_part[i];
        stat[0] = t / (float)Cc;
    }
    __syncthreads();
    const float mu = stat[0];

    float s2 = 0.f;
    for (int i = threadIdx.x; i < Cc; i += 256) { float d = xr[i] - mu; s2 += d * d; }
    #pragma unroll
    for (int o = 16; o > 0; o >>= 1) s2 += __shfl_down(s2, o, 32);
    if (lane == 0) warp_part[wid] = s2;
    __syncthreads();
    if (threadIdx.x == 0) {
        float t = 0.f;
        #pragma unroll
        for (int i = 0; i < 8; i++) t += warp_part[i];
        stat[1] = rsqrtf(t / (float)Cc + 1e-5f);
    }
    __syncthreads();
    const float rstd = stat[1];

    float* orow = out + (size_t)row * Cc;
    for (int i = threadIdx.x; i < Cc; i += 256)
        orow[i] = (xr[i] - mu) * rstd * w[i] + b[i];
}

// ---------------------------------------------------------------------------
// time_shift + lerp mixes
// ---------------------------------------------------------------------------
__global__ void mix3_kernel(const float* __restrict__ ln,
                            const float* __restrict__ mk, const float* __restrict__ mv,
                            const float* __restrict__ mr,
                            bf16* __restrict__ xk, bf16* __restrict__ xv,
                            bf16* __restrict__ xr) {
    int g = blockIdx.x * blockDim.x + threadIdx.x;
    if (g >= Mm * Cc) return;
    int c = g % Cc;
    int row = g / Cc;
    int tt = row & (Tt - 1);
    float cur = ln[g];
    float prev = (tt == 0) ? 0.f : ln[g - Cc];
    float a = mk[c], b = mv[c], r = mr[c];
    xk[g] = (bf16)(cur * a + prev * (1.f - a));
    xv[g] = (bf16)(cur * b + prev * (1.f - b));
    xr[g] = (bf16)(cur * r + prev * (1.f - r));
}

__global__ void mix2_kernel(const float* __restrict__ ln,
                            const float* __restrict__ mk, const float* __restrict__ mr,
                            bf16* __restrict__ xk, bf16* __restrict__ xr) {
    int g = blockIdx.x * blockDim.x + threadIdx.x;
    if (g >= Mm * Cc) return;
    int c = g % Cc;
    int row = g / Cc;
    int tt = row & (Tt - 1);
    float cur = ln[g];
    float prev = (tt == 0) ? 0.f : ln[g - Cc];
    float a = mk[c], r = mr[c];
    xk[g] = (bf16)(cur * a + prev * (1.f - a));
    xr[g] = (bf16)(cur * r + prev * (1.f - r));
}

// ---------------------------------------------------------------------------
// bf16 WMMA GEMM: out[m][n] = epilogue( sum_k A[m][k] * Wt[n][k] )
// 256 thr (8 waves), block tile 128x64, wave tile 32x32 (2x2 of 16x16x32).
// A tile staged via async global->LDS (ASYNCcnt); B tile via TDM (TENSORcnt).
// EPI: 0 = f32 store, 1 = sigmoid f32 store, 2 = relu^2 -> bf16 store
// ---------------------------------------------------------------------------
template <int EPI>
__global__ __launch_bounds__(256)
void gemm_bf16(const bf16* __restrict__ A, const bf16* __restrict__ Wt,
               void* __restrict__ outp, int M, int N, int K) {
    __shared__ bf16 ldsA[TM * LDS_STRIDE];
    __shared__ bf16 ldsB[TN * LDS_STRIDE];

    const int tx   = threadIdx.x;
    const int wid  = tx >> 5;
    const int lane = tx & 31;
    const int half = lane >> 4;     // 0/1: which 16-lane half
    const int ml   = lane & 15;
    const int wm   = wid & 3;       // wave row 0..3
    const int wn   = wid >> 2;      // wave col 0..1
    const int m0blk = blockIdx.y * TM;
    const int n0blk = blockIdx.x * TN;

    v8f acc[2][2];
    #pragma unroll
    for (int i = 0; i < 2; i++)
        #pragma unroll
        for (int j = 0; j < 2; j++)
            acc[i][j] = (v8f){0.f,0.f,0.f,0.f,0.f,0.f,0.f,0.f};

    const int arow  = tx >> 1;           // 0..127
    const int akoff = (tx & 1) * 16;     // 0 or 16 (bf16 elems)
    const int brow  = (tx & 127) >> 1;   // 0..63
#if HAS_TDM
    const unsigned ldsB_off = lds_off(ldsB);
#endif

    for (int k0 = 0; k0 < K; k0 += KC) {
        __syncthreads();

        // ---- stage B tile 64x32 ----
#if HAS_TDM
        if (wid == 0)
            tdm_load_2d(Wt + (size_t)n0blk * K + k0, ldsB_off, TN, N, K);
#elif HAS_ASYNC
        if (tx < 128) {
            const bf16* gsrc = Wt + (size_t)(n0blk + brow) * K + k0 + akoff;
            bf16* ldst = ldsB + brow * LDS_STRIDE + akoff;
            __builtin_amdgcn_global_load_async_to_lds_b128(
                (as1_v4i*)(void*)gsrc, (as3_v4i*)(void*)ldst, 0, 0);
            __builtin_amdgcn_global_load_async_to_lds_b128(
                (as1_v4i*)(void*)(gsrc + 8), (as3_v4i*)(void*)(ldst + 8), 0, 0);
        }
#else
        if (tx < 128) {
            const uint4* src = (const uint4*)(Wt + (size_t)(n0blk + brow) * K + k0 + akoff);
            uint4* dst = (uint4*)(ldsB + brow * LDS_STRIDE + akoff);
            dst[0] = src[0];
            dst[1] = src[1];
        }
#endif

        // ---- stage A tile 128x32 (each thread moves 32B) ----
        {
            const bf16* gsrc = A + (size_t)(m0blk + arow) * K + k0 + akoff;
#if HAS_ASYNC
            bf16* ldst = ldsA + arow * LDS_STRIDE + akoff;
            __builtin_amdgcn_global_load_async_to_lds_b128(
                (as1_v4i*)(void*)gsrc, (as3_v4i*)(void*)ldst, 0, 0);
            __builtin_amdgcn_global_load_async_to_lds_b128(
                (as1_v4i*)(void*)(gsrc + 8), (as3_v4i*)(void*)(ldst + 8), 0, 0);
#else
            const uint4* src = (const uint4*)gsrc;
            uint4* dst = (uint4*)(ldsA + arow * LDS_STRIDE + akoff);
            dst[0] = src[0];
            dst[1] = src[1];
#endif
            if (k0 + KC < K) __builtin_prefetch(gsrc + KC, 0, 2);
        }

#if HAS_ASYNC
        wait_async0();
#endif
#if HAS_TDM
        if (wid == 0) __builtin_amdgcn_s_wait_tensorcnt(0);
#endif
        __syncthreads();

        // A fragments: lanes 0-15 hold K {0..7,16..23}, lanes 16-31 hold {8..15,24..31}
        v16bf af[2];
        #pragma unroll
        for (int i = 0; i < 2; i++) {
            const bf16* rp = ldsA + (wm * 32 + i * 16 + ml) * LDS_STRIDE;
            Frag f;
            f.u[0] = *(const uint4*)(rp + half * 8);
            f.u[1] = *(const uint4*)(rp + 16 + half * 8);
            af[i] = f.v;
        }
        // B fragments: lane n = ml, lanes 0-15 hold K {0..7,16..23}, lanes 16-31 hold {8..15,24..31}
        v16bf bfr[2];
        #pragma unroll
        for (int j = 0; j < 2; j++) {
            const bf16* cp = ldsB + (wn * 32 + j * 16 + ml) * LDS_STRIDE;
            Frag f;
            f.u[0] = *(const uint4*)(cp + half * 8);
            f.u[1] = *(const uint4*)(cp + 16 + half * 8);
            bfr[j] = f.v;
        }
        #pragma unroll
        for (int i = 0; i < 2; i++)
            #pragma unroll
            for (int j = 0; j < 2; j++)
                acc[i][j] = __builtin_amdgcn_wmma_f32_16x16x32_bf16(
                    false, af[i], false, bfr[j], (short)0, acc[i][j], false, false);
    }

    // store: VGPR e, lanes 0-15 -> row m0+e, lanes 16-31 -> row m0+8+e, col = ml
    #pragma unroll
    for (int i = 0; i < 2; i++) {
        #pragma unroll
        for (int j = 0; j < 2; j++) {
            const int mbase = m0blk + wm * 32 + i * 16 + half * 8;
            const int col   = n0blk + wn * 32 + j * 16 + ml;
            #pragma unroll
            for (int e = 0; e < 8; e++) {
                float val = acc[i][j][e];
                size_t o = (size_t)(mbase + e) * N + col;
                if (EPI == 0) {
                    ((float*)outp)[o] = val;
                } else if (EPI == 1) {
                    ((float*)outp)[o] = 1.f / (1.f + __expf(-val));
                } else {
                    float t = fmaxf(val, 0.f);
                    ((bf16*)outp)[o] = (bf16)(t * t);
                }
            }
        }
    }
}

// ---------------------------------------------------------------------------
// WKV recurrence (numerically stable), one thread per (b,c) chain
// ---------------------------------------------------------------------------
__global__ void wkv_kernel(const float* __restrict__ k, const float* __restrict__ v,
                           const float* __restrict__ td, const float* __restrict__ tf,
                           float* __restrict__ y) {
    int g = blockIdx.x * blockDim.x + threadIdx.x;
    if (g >= Bb * Cc) return;
    int c = g % Cc;
    int b = g / Cc;
    const float w = -__expf(td[c]);
    const float u = tf[c];
    float aa = 0.f, bb = 0.f, pp = -1e38f;
    size_t idx = (size_t)b * Tt * Cc + c;
    float kt = k[idx], vt = v[idx];
    for (int t = 0; t < Tt; t++) {
        float kn = 0.f, vn = 0.f;
        if (t + 1 < Tt) { kn = k[idx + Cc]; vn = v[idx + Cc]; }   // prefetch next step
        float ww = u + kt;
        float p  = fmaxf(pp, ww);
        float e1 = __expf(pp - p), e2 = __expf(ww - p);
        y[idx] = (e1 * aa + e2 * vt) / (e1 * bb + e2);
        float ww2 = pp + w;
        float p2  = fmaxf(ww2, kt);
        float e1b = __expf(ww2 - p2), e2b = __expf(kt - p2);
        aa = e1b * aa + e2b * vt;
        bb = e1b * bb + e2b;
        pp = p2;
        kt = kn; vt = vn;
        idx += Cc;
    }
}

// ---------------------------------------------------------------------------
__global__ void ry_kernel(const float* __restrict__ r, const float* __restrict__ y,
                          bf16* __restrict__ out) {
    int g = blockIdx.x * blockDim.x + threadIdx.x;
    if (g >= Mm * Cc) return;
    out[g] = (bf16)(r[g] * y[g]);
}

// ---------------------------------------------------------------------------
// LIF (tau=2, vth=1, hard reset 0) + residual add: xout = xin + lif(a)
// ---------------------------------------------------------------------------
__global__ void lif_add_kernel(const float* __restrict__ xin, const float* __restrict__ a,
                               float* __restrict__ xout) {
    int g = blockIdx.x * blockDim.x + threadIdx.x;
    if (g >= Bb * Cc) return;
    int c = g % Cc;
    int b = g / Cc;
    float vm = 0.f;
    size_t idx = (size_t)b * Tt * Cc + c;
    for (int t = 0; t < Tt; t++) {
        vm = 0.5f * (vm + a[idx]);
        float s = (vm >= 1.f) ? 1.f : 0.f;
        xout[idx] = xin[idx] + s;
        vm = (s > 0.f) ? 0.f : vm;
        idx += Cc;
    }
}

// fused: f = r2*kv ; out = x1 + lif(f)
__global__ void lif2_kernel(const float* __restrict__ x1, const float* __restrict__ kv,
                            const float* __restrict__ r2, float* __restrict__ out) {
    int g = blockIdx.x * blockDim.x + threadIdx.x;
    if (g >= Bb * Cc) return;
    int c = g % Cc;
    int b = g / Cc;
    float vm = 0.f;
    size_t idx = (size_t)b * Tt * Cc + c;
    for (int t = 0; t < Tt; t++) {
        float f = r2[idx] * kv[idx];
        vm = 0.5f * (vm + f);
        float s = (vm >= 1.f) ? 1.f : 0.f;
        out[idx] = x1[idx] + s;
        vm = (s > 0.f) ? 0.f : vm;
        idx += Cc;
    }
}

// ---------------------------------------------------------------------------
static inline size_t al256(size_t x) { return (x + 255) & ~(size_t)255; }

extern "C" void kernel_launch(void* const* d_in, const int* in_sizes, int n_in,
                              void* d_out, int out_size, void* d_ws, size_t ws_size,
                              hipStream_t stream) {
    const float* x        = (const float*)d_in[0];
    const float* ln1_w    = (const float*)d_in[1];
    const float* ln1_b    = (const float*)d_in[2];
    const float* ln2_w    = (const float*)d_in[3];
    const float* ln2_b    = (const float*)d_in[4];
    const float* td       = (const float*)d_in[5];
    const float* tf       = (const float*)d_in[6];
    const float* amk      = (const float*)d_in[7];
    const float* amv      = (const float*)d_in[8];
    const float* amr      = (const float*)d_in[9];
    const float* aWk      = (const float*)d_in[10];
    const float* aWv      = (const float*)d_in[11];
    const float* aWr      = (const float*)d_in[12];
    const float* aWo      = (const float*)d_in[13];
    const float* fmk      = (const float*)d_in[14];
    const float* fmr      = (const float*)d_in[15];
    const float* fWk      = (const float*)d_in[16];
    const float* fWv      = (const float*)d_in[17];
    const float* fWr      = (const float*)d_in[18];
    float* out = (float*)d_out;

    // ---- workspace partition ----
    char* ws = (char*)d_ws;
    size_t off = 0;
    const size_t CC2 = (size_t)Cc * Cc * sizeof(bf16);
    const size_t CH2 = (size_t)Cc * Hh * sizeof(bf16);
    const size_t MC2 = (size_t)Mm * Cc * sizeof(bf16);
    const size_t MC4 = (size_t)Mm * Cc * sizeof(float);
    const size_t MH2 = (size_t)Mm * Hh * sizeof(bf16);

    bf16* wAkT = (bf16*)(ws + off); off = al256(off + CC2);
    bf16* wAvT = (bf16*)(ws + off); off = al256(off + CC2);
    bf16* wArT = (bf16*)(ws + off); off = al256(off + CC2);
    bf16* wAoT = (bf16*)(ws + off); off = al256(off + CC2);
    bf16* wFkT = (bf16*)(ws + off); off = al256(off + CH2);   // [H][C]
    bf16* wFvT = (bf16*)(ws + off); off = al256(off + CH2);   // [C][H]
    bf16* wFrT = (bf16*)(ws + off); off = al256(off + CC2);
    bf16* bufA = (bf16*)(ws + off); off = al256(off + MC2);
    bf16* bufB = (bf16*)(ws + off); off = al256(off + MC2);
    bf16* bufC = (bf16*)(ws + off); off = al256(off + MC2);
    float* buf1 = (float*)(ws + off); off = al256(off + MC4);
    float* buf2 = (float*)(ws + off); off = al256(off + MC4);
    float* buf3 = (float*)(ws + off); off = al256(off + MC4);
    float* buf4 = (float*)(ws + off); off = al256(off + MC4);
    bf16* hbuf = (bf16*)(ws + off); off = al256(off + MH2);
    (void)ws_size; (void)n_in; (void)in_sizes; (void)out_size;

    const int TPB = 256;
    const int gCC = (Cc * Cc + TPB - 1) / TPB;
    const int gCH = (Cc * Hh + TPB - 1) / TPB;
    const int gMC = (Mm * Cc + TPB - 1) / TPB;
    const int gBC = (Bb * Cc + TPB - 1) / TPB;

    // ---- weight packing (f32 [K][N] -> bf16 [N][K]) ----
    transpose_pack_bf16<<<gCC, TPB, 0, stream>>>(aWk, wAkT, Cc, Cc);
    transpose_pack_bf16<<<gCC, TPB, 0, stream>>>(aWv, wAvT, Cc, Cc);
    transpose_pack_bf16<<<gCC, TPB, 0, stream>>>(aWr, wArT, Cc, Cc);
    transpose_pack_bf16<<<gCC, TPB, 0, stream>>>(aWo, wAoT, Cc, Cc);
    transpose_pack_bf16<<<gCH, TPB, 0, stream>>>(fWk, wFkT, Cc, Hh);  // Wk: K=C,N=H
    transpose_pack_bf16<<<gCH, TPB, 0, stream>>>(fWv, wFvT, Hh, Cc);  // Wv: K=H,N=C
    transpose_pack_bf16<<<gCC, TPB, 0, stream>>>(fWr, wFrT, Cc, Cc);

    dim3 gemmGridC(Cc / TN, Mm / TM);   // N=768
    dim3 gemmGridH(Hh / TN, Mm / TM);   // N=3072

    // ---- attention (time-mix) path ----
    ln_kernel<<<Mm, TPB, 0, stream>>>(x, ln1_w, ln1_b, buf1);
    mix3_kernel<<<gMC, TPB, 0, stream>>>(buf1, amk, amv, amr, bufA, bufB, bufC);
    gemm_bf16<0><<<gemmGridC, TPB, 0, stream>>>(bufA, wAkT, buf1, Mm, Cc, Cc);  // k
    gemm_bf16<0><<<gemmGridC, TPB, 0, stream>>>(bufB, wAvT, buf2, Mm, Cc, Cc);  // v
    gemm_bf16<1><<<gemmGridC, TPB, 0, stream>>>(bufC, wArT, buf3, Mm, Cc, Cc);  // r=sigmoid
    wkv_kernel<<<gBC, TPB, 0, stream>>>(buf1, buf2, td, tf, buf4);              // y
    ry_kernel<<<gMC, TPB, 0, stream>>>(buf3, buf4, bufC);                       // r*y -> bf16
    gemm_bf16<0><<<gemmGridC, TPB, 0, stream>>>(bufC, wAoT, buf1, Mm, Cc, Cc);  // a
    lif_add_kernel<<<gBC, TPB, 0, stream>>>(x, buf1, buf2);                     // x1

    // ---- ffn (channel-mix) path ----
    ln_kernel<<<Mm, TPB, 0, stream>>>(buf2, ln2_w, ln2_b, buf3);
    mix2_kernel<<<gMC, TPB, 0, stream>>>(buf3, fmk, fmr, bufA, bufB);
    gemm_bf16<2><<<gemmGridH, TPB, 0, stream>>>(bufA, wFkT, hbuf, Mm, Hh, Cc);  // relu^2 -> bf16
    gemm_bf16<0><<<gemmGridC, TPB, 0, stream>>>(hbuf, wFvT, buf1, Mm, Cc, Hh);  // kv
    gemm_bf16<1><<<gemmGridC, TPB, 0, stream>>>(bufB, wFrT, buf3, Mm, Cc, Cc);  // r2=sigmoid
    lif2_kernel<<<gBC, TPB, 0, stream>>>(buf2, buf1, buf3, out);
}
